// MultiHeadAttention_60327110639894
// MI455X (gfx1250) — compile-verified, matrix operand order checked by reading, not run on hardware
//
#include <hip/hip_runtime.h>

// ---------------------------------------------------------------------------
// MHA forward for MI455X (gfx1250, wave32, WMMA 16x16x32 f16 -> f32 acc)
// TDM (tensor_load_to_lds) double-buffered K/V staging in the attention loop.
// ---------------------------------------------------------------------------

typedef __attribute__((ext_vector_type(16))) _Float16     v16h;
typedef __attribute__((ext_vector_type(8)))  float        v8f;
typedef __attribute__((ext_vector_type(4)))  unsigned int v4u;
typedef __attribute__((ext_vector_type(8)))  int          v8i;
typedef __attribute__((ext_vector_type(4)))  int          v4i;

#define D_MODEL 1024
#define NHEAD   16
#define DK      64
#define BATCH   4
#define SEQ     2048
#define MTOT    (BATCH * SEQ)   // 8192 rows

#if defined(__has_builtin)
#if __has_builtin(__builtin_amdgcn_tensor_load_to_lds) && \
    __has_builtin(__builtin_amdgcn_s_wait_tensorcnt)
#define HAVE_TDM 1
#endif
#endif
#ifndef HAVE_TDM
#define HAVE_TDM 0
#endif

union AFrag { v16h v; uint4 q[2]; };

// Load one 16x32 f16 A/B fragment. `p` points at this lane's first 8-element
// chunk (row*stride + k0 + half*8); second chunk lives 16 elements further.
__device__ __forceinline__ v16h load_frag(const _Float16* p) {
    AFrag f;
    f.q[0] = *(const uint4*)(p);
    f.q[1] = *(const uint4*)(p + 16);
    return f.v;
}

#if HAVE_TDM
// Issue a TDM 2D tile load (f16 elements) into LDS. D# layout per CDNA5 ISA
// §8.3/8.4: group0 = {count=1, lds_addr, global_addr, type=2},
// group1 = {data_size=2B, tensor dims, tile dims, dim0 stride}.
__device__ __forceinline__ void tdm_load_2d(const _Float16* gptr, unsigned ldsOff,
                                            unsigned tensor_d0, unsigned tensor_d1,
                                            unsigned tile_d0, unsigned tile_d1,
                                            unsigned long long stride0) {
    const unsigned long long ga = (unsigned long long)(uintptr_t)gptr;
    v4u g0;
    g0[0] = 1u;                                            // count=1 (valid D#)
    g0[1] = ldsOff;                                        // lds_addr
    g0[2] = (unsigned)ga;                                  // global_addr[31:0]
    g0[3] = (unsigned)((ga >> 32) & 0x01FFFFFFull)         // global_addr[56:32]
            | (2u << 30);                                  // type=2 ("image")
    v8i g1;
    g1[0] = (int)(1u << 16);                               // data_size=1 -> 2 bytes
    g1[1] = (int)((tensor_d0 & 0xFFFFu) << 16);            // tensor_dim0[15:0]
    g1[2] = (int)(((tensor_d0 >> 16) & 0xFFFFu) |
                  ((tensor_d1 & 0xFFFFu) << 16));          // dim0[31:16] | dim1[15:0]
    g1[3] = (int)(((tensor_d1 >> 16) & 0xFFFFu) |
                  ((tile_d0 & 0xFFFFu) << 16));            // dim1[31:16] | tile_dim0
    g1[4] = (int)(tile_d1 & 0xFFFFu);                      // tile_dim1 (tile_dim2=0)
    g1[5] = (int)(stride0 & 0xFFFFFFFFull);                // tensor_dim0_stride[31:0]
    g1[6] = (int)((stride0 >> 32) & 0xFFFFull);            // stride[47:32]
    g1[7] = 0;
    v4i gz = {0, 0, 0, 0};
#if __clang_major__ >= 23
    v8i gz8 = {0, 0, 0, 0, 0, 0, 0, 0};
    __builtin_amdgcn_tensor_load_to_lds(g0, g1, gz, gz, gz8, 0);
#else
    __builtin_amdgcn_tensor_load_to_lds(g0, g1, gz, gz, 0);
#endif
}
#endif  // HAVE_TDM

// ---------------------------------------------------------------------------
// f32 -> f16 conversion (memory bound, trivial vs GEMM time)
// ---------------------------------------------------------------------------
__global__ void cvt_f32_f16(const float* __restrict__ in,
                            _Float16* __restrict__ out, int n) {
    int i = blockIdx.x * blockDim.x + threadIdx.x;
    int stride = gridDim.x * blockDim.x;
    for (; i < n; i += stride) out[i] = (_Float16)in[i];
}

// ---------------------------------------------------------------------------
// GEMM: Y = X (M x 1024) @ W^T (W is 1024 x 1024, row-major N x K) + bias.
// Block: 256 threads = 8 waves, tile 128(M) x 128(N); wave tile 32 x 64.
// All operands stay L2-resident (192MB L2), so fragments come straight from
// global; the loop is WMMA-issue-bound (8 WMMAs / 12 b128 loads per K-step).
// MODE 0: fp32 plain output (final projection, to d_out)
// MODE 1: f16 split-head output  [b,h,s,d]   (Q with scale, K)
// MODE 2: f16 split-head transposed [b,h,d,s] (V)
// ---------------------------------------------------------------------------
template <int MODE>
__global__ __launch_bounds__(256)
void gemm_f16(const _Float16* __restrict__ A, const _Float16* __restrict__ W,
              const float* __restrict__ bias, void* __restrict__ out,
              float scale) {
    const int lane = threadIdx.x & 31;
    const int wave = threadIdx.x >> 5;
    const int r    = lane & 15;
    const int half = lane >> 4;
    const int wm   = wave & 3;          // 4 waves along M
    const int wn   = wave >> 2;         // 2 waves along N
    const int m0   = blockIdx.x * 128 + wm * 32;
    const int n0   = blockIdx.y * 128 + wn * 64;

    v8f acc[2][4] = {};

    for (int kk = 0; kk < D_MODEL; kk += 32) {
        if (kk + 32 < D_MODEL) {
            __builtin_prefetch(A + (size_t)(m0 + r) * D_MODEL + kk + 32, 0, 0);
            __builtin_prefetch(W + (size_t)(n0 + r) * D_MODEL + kk + 32, 0, 0);
        }
        v16h af[2], bf[4];
#pragma unroll
        for (int mi = 0; mi < 2; ++mi)
            af[mi] = load_frag(A + (size_t)(m0 + mi * 16 + r) * D_MODEL + kk + half * 8);
#pragma unroll
        for (int ni = 0; ni < 4; ++ni)
            bf[ni] = load_frag(W + (size_t)(n0 + ni * 16 + r) * D_MODEL + kk + half * 8);
#pragma unroll
        for (int mi = 0; mi < 2; ++mi)
#pragma unroll
            for (int ni = 0; ni < 4; ++ni)
                acc[mi][ni] = __builtin_amdgcn_wmma_f32_16x16x32_f16(
                    false, af[mi], false, bf[ni], (short)0, acc[mi][ni], false, false);
    }

#pragma unroll
    for (int mi = 0; mi < 2; ++mi) {
#pragma unroll
        for (int ni = 0; ni < 4; ++ni) {
            const int col = n0 + ni * 16 + r;
            const float bv = bias[col];
#pragma unroll
            for (int v = 0; v < 8; ++v) {
                const int row = m0 + mi * 16 + v + 8 * half;  // C layout
                const float val = (acc[mi][ni][v] + bv) * scale;
                if (MODE == 0) {
                    ((float*)out)[(size_t)row * D_MODEL + col] = val;
                } else {
                    const int b = row >> 11, s = row & (SEQ - 1);
                    const int h = col >> 6,  d = col & (DK - 1);
                    if (MODE == 1)
                        ((_Float16*)out)[(((size_t)(b * NHEAD + h)) * SEQ + s) * DK + d] =
                            (_Float16)val;
                    else
                        ((_Float16*)out)[(((size_t)(b * NHEAD + h)) * DK + d) * SEQ + s] =
                            (_Float16)val;
                }
            }
        }
    }
}

// ---------------------------------------------------------------------------
// Flash attention: softmax(Q K^T / sqrt(dk)) V per (b,h). Q pre-scaled.
// Grid: (SEQ/128, B*H); 256 threads = 8 waves, each wave owns 16 query rows.
// K / V^T tiles (64 keys) double-buffered in LDS; wave 0 issues the next
// tile's TDM load while the block computes on the current one, so the tile
// DMA rides entirely under the 16 WMMAs + softmax of each iteration.
// ---------------------------------------------------------------------------
__global__ __launch_bounds__(256)
void attn_kernel(const _Float16* __restrict__ Qh, const _Float16* __restrict__ Kh,
                 const _Float16* __restrict__ Vt, _Float16* __restrict__ AO) {
    __shared__ __align__(16) _Float16 Kbuf[2 * 64 * 64];   // [buf][key][d]
    __shared__ __align__(16) _Float16 Vbuf[2 * 64 * 64];   // [buf][d][key]
    __shared__ __align__(16) _Float16 Pt[8][16 * 64];      // per-wave P tile

    const int lane = threadIdx.x & 31;
    const int wave = threadIdx.x >> 5;
    const int r    = lane & 15;
    const int half = lane >> 4;
    const int bh   = blockIdx.y;
    const int q0   = blockIdx.x * 128;
    const int NKT  = SEQ / 64;

    // Q fragments (16 rows x 64, two k-steps), resident for whole kernel.
    const _Float16* qbase = Qh + ((size_t)bh * SEQ + q0 + wave * 16 + r) * DK;
    v16h qf[2];
#pragma unroll
    for (int ks = 0; ks < 2; ++ks)
        qf[ks] = load_frag(qbase + ks * 32 + half * 8);

    v8f oacc[4] = {};
    float rowm[8], rowl[8];
#pragma unroll
    for (int v = 0; v < 8; ++v) { rowm[v] = -3.0e38f; rowl[v] = 0.f; }

    const _Float16* Kg0 = Kh + (size_t)bh * SEQ * DK;   // [s][d], contiguous
    const _Float16* Vg0 = Vt + (size_t)bh * DK * SEQ;   // [d][s], stride SEQ

#if HAVE_TDM
    const unsigned kLds0 = (unsigned)(unsigned long long)(uintptr_t)(&Kbuf[0]);
    const unsigned kLds1 = (unsigned)(unsigned long long)(uintptr_t)(&Kbuf[64 * 64]);
    const unsigned vLds0 = (unsigned)(unsigned long long)(uintptr_t)(&Vbuf[0]);
    const unsigned vLds1 = (unsigned)(unsigned long long)(uintptr_t)(&Vbuf[64 * 64]);
    if (wave == 0) {  // prologue: tile 0 -> buffer 0
        tdm_load_2d(Kg0, kLds0, 64, 64, 64, 64, 64ull);
        tdm_load_2d(Vg0, vLds0, 64, 64, 64, 64, (unsigned long long)SEQ);
    }
#else
    const int t  = threadIdx.x;
    const int rr = t >> 2;           // 0..63 tile row
    const int cc = (t & 3) << 4;     // 0,16,32,48
#endif

    for (int kt = 0; kt < NKT; ++kt) {
#if HAVE_TDM
        if (wave == 0) {
            if (kt + 1 < NKT) {  // issue next tile into the other buffer
                const int nb = (kt + 1) & 1;
                tdm_load_2d(Kg0 + (size_t)(kt + 1) * 64 * DK,
                            nb ? kLds1 : kLds0, 64, 64, 64, 64, 64ull);
                tdm_load_2d(Vg0 + (size_t)(kt + 1) * 64,
                            nb ? vLds1 : vLds0, 64, 64, 64, 64,
                            (unsigned long long)SEQ);
                __builtin_amdgcn_s_wait_tensorcnt((short)2);  // current tile done
            } else {
                __builtin_amdgcn_s_wait_tensorcnt((short)0);
            }
        }
        __syncthreads();  // current tile visible to all waves
#else
        __syncthreads();
        {
            const int k0 = kt * 64;
            const int bo = (kt & 1) * 64 * 64;
            const _Float16* kg = Kg0 + (size_t)(k0 + rr) * DK + cc;
            *(uint4*)&Kbuf[bo + rr * 64 + cc]     = *(const uint4*)kg;
            *(uint4*)&Kbuf[bo + rr * 64 + cc + 8] = *(const uint4*)(kg + 8);
            const _Float16* vg = Vg0 + (size_t)rr * SEQ + k0 + cc;
            *(uint4*)&Vbuf[bo + rr * 64 + cc]     = *(const uint4*)vg;
            *(uint4*)&Vbuf[bo + rr * 64 + cc + 8] = *(const uint4*)(vg + 8);
        }
        __syncthreads();
#endif
        const _Float16* Kc = &Kbuf[(kt & 1) * 64 * 64];
        const _Float16* Vc = &Vbuf[(kt & 1) * 64 * 64];

        // S = Q K^T : 16 x 64 scores in 4 accumulators (8 WMMAs).
        v8f sc[4] = {};
#pragma unroll
        for (int ni = 0; ni < 4; ++ni)
#pragma unroll
            for (int ks = 0; ks < 2; ++ks) {
                v16h kf = load_frag(&Kc[(ni * 16 + r) * 64 + ks * 32 + half * 8]);
                sc[ni] = __builtin_amdgcn_wmma_f32_16x16x32_f16(
                    false, qf[ks], false, kf, (short)0, sc[ni], false, false);
            }

        // Online softmax. Row v (+8*half) lives across this lane's 16-lane
        // half-group -> xor-shuffle reduce with masks 1,2,4,8.
        float mx[8];
#pragma unroll
        for (int v = 0; v < 8; ++v)
            mx[v] = fmaxf(fmaxf(sc[0][v], sc[1][v]), fmaxf(sc[2][v], sc[3][v]));
#pragma unroll
        for (int m = 1; m <= 8; m <<= 1)
#pragma unroll
            for (int v = 0; v < 8; ++v)
                mx[v] = fmaxf(mx[v], __shfl_xor(mx[v], m, 32));

        float alpha[8], rs[8];
#pragma unroll
        for (int v = 0; v < 8; ++v) {
            const float mn = fmaxf(rowm[v], mx[v]);
            alpha[v] = __expf(rowm[v] - mn);
            rowm[v]  = mn;
            rs[v]    = 0.f;
        }
#pragma unroll
        for (int ni = 0; ni < 4; ++ni)
#pragma unroll
            for (int v = 0; v < 8; ++v) {
                const float p = __expf(sc[ni][v] - rowm[v]);
                rs[v] += p;
                Pt[wave][(v + 8 * half) * 64 + ni * 16 + r] = (_Float16)p;
            }
#pragma unroll
        for (int m = 1; m <= 8; m <<= 1)
#pragma unroll
            for (int v = 0; v < 8; ++v)
                rs[v] += __shfl_xor(rs[v], m, 32);
#pragma unroll
        for (int v = 0; v < 8; ++v)
            rowl[v] = rowl[v] * alpha[v] + rs[v];
#pragma unroll
        for (int dt = 0; dt < 4; ++dt)
#pragma unroll
            for (int v = 0; v < 8; ++v)
                oacc[dt][v] *= alpha[v];

        // O += P V. No block barrier needed: the P tile is per-wave and LDS
        // ops from one wave are architecturally in-order (DScnt ordering).
#pragma unroll
        for (int ks = 0; ks < 2; ++ks) {
            v16h pf = load_frag(&Pt[wave][r * 64 + ks * 32 + half * 8]);
#pragma unroll
            for (int dt = 0; dt < 4; ++dt) {
                v16h vf = load_frag(&Vc[(dt * 16 + r) * 64 + ks * 32 + half * 8]);
                oacc[dt] = __builtin_amdgcn_wmma_f32_16x16x32_f16(
                    false, pf, false, vf, (short)0, oacc[dt], false, false);
            }
        }
        __syncthreads();  // tile consumed; safe to overwrite this buffer
    }

    // Normalize and write merged-head f16 output (B*S, 1024) for Wo GEMM.
    const int b = bh >> 4, h = bh & 15;
#pragma unroll
    for (int dt = 0; dt < 4; ++dt)
#pragma unroll
        for (int v = 0; v < 8; ++v) {
            const int row = q0 + wave * 16 + v + 8 * half;
            const int col = h * DK + dt * 16 + r;
            AO[((size_t)(b * SEQ + row)) * D_MODEL + col] =
                (_Float16)(oacc[dt][v] / rowl[v]);
        }
}

// ---------------------------------------------------------------------------
// Host orchestration
// ---------------------------------------------------------------------------
extern "C" void kernel_launch(void* const* d_in, const int* in_sizes, int n_in,
                              void* d_out, int out_size, void* d_ws, size_t ws_size,
                              hipStream_t stream) {
    (void)in_sizes; (void)n_in; (void)out_size; (void)ws_size;

    const float* query = (const float*)d_in[0];
    const float* key   = (const float*)d_in[1];
    const float* value = (const float*)d_in[2];
    const float* Wq    = (const float*)d_in[3];
    const float* bq    = (const float*)d_in[4];
    const float* Wk    = (const float*)d_in[5];
    const float* bk    = (const float*)d_in[6];
    const float* Wv    = (const float*)d_in[7];
    const float* bv    = (const float*)d_in[8];
    const float* Wo    = (const float*)d_in[9];
    const float* bo    = (const float*)d_in[10];

    // Workspace layout (f16): 16MB act buffer (reused) + 4x2MB weights +
    // Qh/Kh/Vt 16MB each = 72MB total.
    char* w = (char*)d_ws;
    const size_t ACT_BYTES = (size_t)MTOT * D_MODEL * sizeof(_Float16);
    const size_t W_BYTES   = (size_t)D_MODEL * D_MODEL * sizeof(_Float16);
    _Float16* bufX = (_Float16*)(w);                       // activations (reused)
    _Float16* Wq16 = (_Float16*)(w + ACT_BYTES);
    _Float16* Wk16 = (_Float16*)(w + ACT_BYTES + 1 * W_BYTES);
    _Float16* Wv16 = (_Float16*)(w + ACT_BYTES + 2 * W_BYTES);
    _Float16* Wo16 = (_Float16*)(w + ACT_BYTES + 3 * W_BYTES);
    _Float16* Qh   = (_Float16*)(w + ACT_BYTES + 4 * W_BYTES);
    _Float16* Kh   = (_Float16*)(w + 2 * ACT_BYTES + 4 * W_BYTES);
    _Float16* Vt   = (_Float16*)(w + 3 * ACT_BYTES + 4 * W_BYTES);

    const int nAct = MTOT * D_MODEL;          // 8388608
    const int nW   = D_MODEL * D_MODEL;       // 1048576
    const dim3 cvtGrid(2048), cvtBlk(256);
    const dim3 gemmGrid(MTOT / 128, D_MODEL / 128), gemmBlk(256);
    const dim3 attnGrid(SEQ / 128, BATCH * NHEAD), attnBlk(256);
    const float qscale = 0.125f;              // 1/sqrt(d_k)

    // Weights to f16 (once per launch).
    cvt_f32_f16<<<cvtGrid, cvtBlk, 0, stream>>>(Wq, Wq16, nW);
    cvt_f32_f16<<<cvtGrid, cvtBlk, 0, stream>>>(Wk, Wk16, nW);
    cvt_f32_f16<<<cvtGrid, cvtBlk, 0, stream>>>(Wv, Wv16, nW);
    cvt_f32_f16<<<cvtGrid, cvtBlk, 0, stream>>>(Wo, Wo16, nW);

    // Q projection (scale folded in), split-head layout.
    cvt_f32_f16<<<cvtGrid, cvtBlk, 0, stream>>>(query, bufX, nAct);
    gemm_f16<1><<<gemmGrid, gemmBlk, 0, stream>>>(bufX, Wq16, bq, Qh, qscale);
    // K projection, split-head layout.
    cvt_f32_f16<<<cvtGrid, cvtBlk, 0, stream>>>(key, bufX, nAct);
    gemm_f16<1><<<gemmGrid, gemmBlk, 0, stream>>>(bufX, Wk16, bk, Kh, 1.0f);
    // V projection, transposed split-head layout [b,h,d,s].
    cvt_f32_f16<<<cvtGrid, cvtBlk, 0, stream>>>(value, bufX, nAct);
    gemm_f16<2><<<gemmGrid, gemmBlk, 0, stream>>>(bufX, Wv16, bv, Vt, 1.0f);

    // Attention -> merged heads f16, reusing bufX.
    attn_kernel<<<attnGrid, attnBlk, 0, stream>>>(Qh, Kh, Vt, bufX);

    // Output projection, fp32 result.
    gemm_f16<0><<<gemmGrid, gemmBlk, 0, stream>>>(bufX, Wo16, bo, d_out, 1.0f);
}
